// LoRAAttention_37890201485926
// MI455X (gfx1250) — matrix operand order repaired and checked by reading, not compiled, not run on hardware
//
#include <hip/hip_runtime.h>

typedef __attribute__((ext_vector_type(16))) _Float16 v16h;
typedef __attribute__((ext_vector_type(8)))  _Float16 v8h;
typedef __attribute__((ext_vector_type(4)))  _Float16 v4h;
typedef __attribute__((ext_vector_type(8)))  float    v8f;
typedef __attribute__((ext_vector_type(4)))  float    v4f;

#define WMMA_F16(a,b,c) __builtin_amdgcn_wmma_f32_16x16x32_f16(false,(a),false,(b),(short)0,(c),false,false)

// Async global->LDS copy, 16B per lane, tracked by ASYNCcnt (CDNA5 §10 async ops).
// lds: wave-relative LDS byte address (low 32 bits of generic shared pointer);
// gp:  64-bit global address in a VGPR pair.
#define ASYNC_LOAD_B128(lds, gp)                                        \
    asm volatile("global_load_async_to_lds_b128 %0, %1, off"            \
                 :: "v"(lds), "v"(gp) : "memory")
#define WAIT_ASYNC(n) asm volatile("s_wait_asynccnt " #n ::: "memory")

__device__ __forceinline__ unsigned lds_addr(const void* p) {
    return (unsigned)(uintptr_t)p;   // flat->LDS aperture keeps offset in addr[31:0]
}

// A-fragment (16x32 f16, MxK): lane m=lane&15, kb=(lane>>4)*8; K in {kb..kb+7, kb+16..kb+23}
__device__ __forceinline__ v16h load_fragA(const _Float16* base, int ld, int lane) {
    const int m  = lane & 15;
    const int kb = (lane >> 4) << 3;
    const _Float16* p = base + m * ld + kb;
    v8h lo = *(const v8h*)p;
    v8h hi = *(const v8h*)(p + 16);
    v16h r;
#pragma unroll
    for (int i = 0; i < 8; i++) { r[i] = lo[i]; r[i + 8] = hi[i]; }
    return r;
}

// B-fragment (32x16 f16, KxN) from N-major (Bt[n][k]) storage:
// lane n=lane&15 holds column n, kb=(lane>>4)*8; K in {kb..kb+7, kb+16..kb+23}
__device__ __forceinline__ v16h load_fragB(const _Float16* base, int ld, int lane) {
    const int n  = lane & 15;
    const int kb = (lane >> 4) << 3;
    const _Float16* p = base + n * ld + kb;
    v8h lo = *(const v8h*)p;
    v8h hi = *(const v8h*)(p + 16);
    v16h r;
#pragma unroll
    for (int i = 0; i < 8; i++) { r[i] = lo[i]; r[i + 8] = hi[i]; }
    return r;
}

// ---------------- Kernel 0: x (f32) -> f16 ----------------
__global__ __launch_bounds__(256) void k_cvt_x(const float* __restrict__ x,
                                               _Float16* __restrict__ xh) {
    int i = blockIdx.x * 256 + threadIdx.x;   // over 8M/4 float4s (exact)
    v4f f = *(const v4f*)(x + 4 * (size_t)i);
    v4h h;
#pragma unroll
    for (int e = 0; e < 4; e++) h[e] = (_Float16)f[e];
    *(v4h*)(xh + 4 * (size_t)i) = h;
}

// ---------------- Kernel 1: fold LoRA into qkv_w, convert weights to f16 ----------------
__global__ __launch_bounds__(256) void k_fuse_w(const float* __restrict__ qkv_w,
                                                const float* __restrict__ lqa,
                                                const float* __restrict__ lqb,
                                                const float* __restrict__ lva,
                                                const float* __restrict__ lvb,
                                                const float* __restrict__ projw,
                                                _Float16* __restrict__ wf,
                                                _Float16* __restrict__ pwh) {
    int idx = blockIdx.x * 256 + threadIdx.x;       // 3M elements (exact)
    int j = idx >> 10, k = idx & 1023;
    float w = qkv_w[idx];
    if (j < 1024) {                                 // q rows: + 2 * B_q A_q
        float a = 0.f;
#pragma unroll
        for (int r = 0; r < 8; r++) a += lqb[j * 8 + r] * lqa[r * 1024 + k];
        w += 2.0f * a;
    } else if (j >= 2048) {                         // v rows: + 2 * B_v A_v
        int jj = j - 2048;
        float a = 0.f;
#pragma unroll
        for (int r = 0; r < 8; r++) a += lvb[jj * 8 + r] * lva[r * 1024 + k];
        w += 2.0f * a;
    }
    wf[idx] = (_Float16)w;
    if (idx < 1024 * 1024) pwh[idx] = (_Float16)projw[idx];
}

// ---------------- Kernel 2: QKV GEMM (8192x3072x1024), async double-buffered ----------------
__global__ __launch_bounds__(256) void k_qkv(const _Float16* __restrict__ xh,
                                             const _Float16* __restrict__ wf,
                                             const float* __restrict__ bias,
                                             _Float16* __restrict__ q,
                                             _Float16* __restrict__ k,
                                             _Float16* __restrict__ v) {
    __shared__ __align__(16) _Float16 sA[2][128][40];
    __shared__ __align__(16) _Float16 sB[2][128][40];
    const int t = threadIdx.x, lane = t & 31, w = t >> 5;
    const int wm = w >> 2, wn = w & 3;
    const int m0 = blockIdx.y * 128, n0 = blockIdx.x * 128;
    const int lrow = t >> 1, lcg = (t & 1) * 16;

    const _Float16* gaBase = xh + (size_t)(m0 + lrow) * 1024 + lcg;
    const _Float16* gbBase = wf + (size_t)(n0 + lrow) * 1024 + lcg;

    auto issue = [&](int kt, int buf) {
        const _Float16* ga = gaBase + kt * 32;
        const _Float16* gb = gbBase + kt * 32;
        unsigned la = lds_addr(&sA[buf][lrow][lcg]);
        unsigned lb = lds_addr(&sB[buf][lrow][lcg]);
        ASYNC_LOAD_B128(la,      ga);
        ASYNC_LOAD_B128(la + 16, ga + 8);
        ASYNC_LOAD_B128(lb,      gb);
        ASYNC_LOAD_B128(lb + 16, gb + 8);
    };

    v8f acc[4][2];
#pragma unroll
    for (int i = 0; i < 4; i++)
#pragma unroll
        for (int j = 0; j < 2; j++) acc[i][j] = v8f{};

    issue(0, 0);
    for (int kt = 0; kt < 32; ++kt) {
        const int cur = kt & 1;
        if (kt < 31) { issue(kt + 1, 1 - cur); WAIT_ASYNC(0x4); }
        else         { WAIT_ASYNC(0x0); }
        __syncthreads();
        v16h af[4], bf[2];
#pragma unroll
        for (int i = 0; i < 4; i++) af[i] = load_fragA(&sA[cur][wm * 64 + i * 16][0], 40, lane);
#pragma unroll
        for (int j = 0; j < 2; j++) bf[j] = load_fragB(&sB[cur][wn * 32 + j * 16][0], 40, lane);
#pragma unroll
        for (int i = 0; i < 4; i++)
#pragma unroll
            for (int j = 0; j < 2; j++) acc[i][j] = WMMA_F16(af[i], bf[j], acc[i][j]);
        __syncthreads();   // all waves done with buf cur before it is refilled
    }

    const int mh = (lane >> 4) << 3, nn = lane & 15;
#pragma unroll
    for (int i = 0; i < 4; i++) {
#pragma unroll
        for (int j = 0; j < 2; j++) {
            int gn = n0 + wn * 32 + j * 16 + nn;           // 0..3071
            float bv = bias[gn];
            int s = gn >> 10, h = (gn >> 6) & 15, d = gn & 63;
            _Float16* dst = (s == 0) ? q : (s == 1) ? k : v;
#pragma unroll
            for (int r = 0; r < 8; r++) {
                int gm = m0 + wm * 64 + i * 16 + r + mh;   // 0..8191
                int b = gm >> 10, n = gm & 1023;
                dst[(size_t)((b * 16 + h) * 1024 + n) * 64 + d] = (_Float16)(acc[i][j][r] + bv);
            }
        }
    }
}

// ---------------- Kernel 3: flash attention, one (b,h) x 128-row tile per block ----------------
__global__ __launch_bounds__(256) void k_attn(const _Float16* __restrict__ q,
                                              const _Float16* __restrict__ kk,
                                              const _Float16* __restrict__ vv,
                                              _Float16* __restrict__ o) {
    __shared__ __align__(16) _Float16 sQ[128][72];
    __shared__ __align__(16) _Float16 sK[64][72];
    __shared__ __align__(16) _Float16 sVt[64][72];    // V transposed: sVt[d][j]
    __shared__ __align__(16) _Float16 sP[8][16][72];  // per-wave P tile
    const int t = threadIdx.x, lane = t & 31, w = t >> 5;
    const int bh = blockIdx.y, m0 = blockIdx.x * 128;
    const size_t baseQ = (size_t)bh * 1024 * 64;

    {   // stage Q tile (128x64) via async copy
        int row = t >> 1, cg = (t & 1) * 32;
        const _Float16* g = q + baseQ + (size_t)(m0 + row) * 64 + cg;
        unsigned lq = lds_addr(&sQ[row][cg]);
#pragma unroll
        for (int e = 0; e < 4; e++) ASYNC_LOAD_B128(lq + 16 * e, g + 8 * e);
    }
    WAIT_ASYNC(0x0);
    __syncthreads();
    v16h qf[2];
#pragma unroll
    for (int kc = 0; kc < 2; kc++) qf[kc] = load_fragA(&sQ[w * 16][kc * 32], 72, lane);

    v8f oacc[4];
#pragma unroll
    for (int i = 0; i < 4; i++) oacc[i] = v8f{};
    float mrun[8], lrun[8];
#pragma unroll
    for (int r = 0; r < 8; r++) { mrun[r] = -1e30f; lrun[r] = 0.f; }

    for (int jt = 0; jt < 16; ++jt) {
        __syncthreads();
        {   // stage K async (row-major) and V synchronously (transposed on LDS store)
            int row = t >> 2, cg = (t & 3) * 16;
            const _Float16* gk = kk + baseQ + (size_t)(jt * 64 + row) * 64 + cg;
            unsigned lk = lds_addr(&sK[row][cg]);
            ASYNC_LOAD_B128(lk,      gk);
            ASYNC_LOAD_B128(lk + 16, gk + 8);
            const _Float16* gv = vv + baseQ + (size_t)(jt * 64 + row) * 64 + cg;
            v8h a = *(const v8h*)gv, b2 = *(const v8h*)(gv + 8);
#pragma unroll
            for (int e = 0; e < 8; e++) { sVt[cg + e][row] = a[e]; sVt[cg + 8 + e][row] = b2[e]; }
        }
        WAIT_ASYNC(0x0);
        __syncthreads();

        // S = Q Kt (16 rows x 64 cols per wave), f32 accum
        v8f sacc[4];
#pragma unroll
        for (int nb = 0; nb < 4; nb++) {
            sacc[nb] = v8f{};
#pragma unroll
            for (int kc = 0; kc < 2; kc++) {
                v16h bf = load_fragB(&sK[nb * 16][kc * 32], 72, lane);
                sacc[nb] = WMMA_F16(qf[kc], bf, sacc[nb]);
            }
        }
#pragma unroll
        for (int nb = 0; nb < 4; nb++)
#pragma unroll
            for (int r = 0; r < 8; r++) sacc[nb][r] *= 0.125f;   // D^-0.5

        // online softmax: rows live per-VGPR, cols per-lane -> xor-shuffles within 16-lane halves
        float pv[4][8];
#pragma unroll
        for (int r = 0; r < 8; r++) {
            float mx = fmaxf(fmaxf(sacc[0][r], sacc[1][r]), fmaxf(sacc[2][r], sacc[3][r]));
#pragma unroll
            for (int msk = 1; msk < 16; msk <<= 1) mx = fmaxf(mx, __shfl_xor(mx, msk));
            float mnew  = fmaxf(mrun[r], mx);
            float alpha = __expf(mrun[r] - mnew);
            mrun[r] = mnew;
            float rs = 0.f;
#pragma unroll
            for (int nb = 0; nb < 4; nb++) { float p = __expf(sacc[nb][r] - mnew); pv[nb][r] = p; rs += p; }
#pragma unroll
            for (int msk = 1; msk < 16; msk <<= 1) rs += __shfl_xor(rs, msk);
            lrun[r] = lrun[r] * alpha + rs;
#pragma unroll
            for (int nb = 0; nb < 4; nb++) oacc[nb][r] *= alpha;
        }

        // C-layout -> A-fragment layout via per-wave LDS tile
        const int mh = (lane >> 4) << 3, nn = lane & 15;
#pragma unroll
        for (int nb = 0; nb < 4; nb++)
#pragma unroll
            for (int r = 0; r < 8; r++) sP[w][r + mh][nb * 16 + nn] = (_Float16)pv[nb][r];
        __syncthreads();

        // O += P V
#pragma unroll
        for (int kc = 0; kc < 2; kc++) {
            v16h af = load_fragA(&sP[w][0][kc * 32], 72, lane);
#pragma unroll
            for (int db = 0; db < 4; db++) {
                v16h bf = load_fragB(&sVt[db * 16][kc * 32], 72, lane);
                oacc[db] = WMMA_F16(af, bf, oacc[db]);
            }
        }
    }

    const int mh = (lane >> 4) << 3, nn = lane & 15;
#pragma unroll
    for (int db = 0; db < 4; db++)
#pragma unroll
        for (int r = 0; r < 8; r++) {
            int row = m0 + w * 16 + r + mh;
            o[baseQ + (size_t)row * 64 + db * 16 + nn] = (_Float16)(oacc[db][r] / lrun[r]);
        }
}

// ---------------- Kernel 4: output projection (8192x1024x1024) + bias, async double-buffered ----------------
__global__ __launch_bounds__(256) void k_proj(const _Float16* __restrict__ ows,
                                              const _Float16* __restrict__ pw,
                                              const float* __restrict__ pb,
                                              float* __restrict__ out) {
    __shared__ __align__(16) _Float16 sA[2][128][40];
    __shared__ __align__(16) _Float16 sB[2][128][40];
    const int t = threadIdx.x, lane = t & 31, w = t >> 5;
    const int wm = w >> 2, wn = w & 3;
    const int m0 = blockIdx.y * 128, n0 = blockIdx.x * 128;
    const int lrow = t >> 1, lcg = (t & 1) * 16;

    const int gm_l = m0 + lrow;
    const int b_l = gm_l >> 10, n_l = gm_l & 1023;
    const _Float16* gbBase = pw + (size_t)(n0 + lrow) * 1024 + lcg;

    auto issue = [&](int kt, int buf) {
        int h = kt >> 1, d0 = (kt & 1) * 32;   // A col c = h*64+d, gathered from [BH][N][64]
        const _Float16* ga = ows + (size_t)((b_l * 16 + h) * 1024 + n_l) * 64 + d0 + lcg;
        const _Float16* gb = gbBase + kt * 32;
        unsigned la = lds_addr(&sA[buf][lrow][lcg]);
        unsigned lb = lds_addr(&sB[buf][lrow][lcg]);
        ASYNC_LOAD_B128(la,      ga);
        ASYNC_LOAD_B128(la + 16, ga + 8);
        ASYNC_LOAD_B128(lb,      gb);
        ASYNC_LOAD_B128(lb + 16, gb + 8);
    };

    v8f acc[4][2];
#pragma unroll
    for (int i = 0; i < 4; i++)
#pragma unroll
        for (int j = 0; j < 2; j++) acc[i][j] = v8f{};

    issue(0, 0);
    for (int kt = 0; kt < 32; ++kt) {
        const int cur = kt & 1;
        if (kt < 31) { issue(kt + 1, 1 - cur); WAIT_ASYNC(0x4); }
        else         { WAIT_ASYNC(0x0); }
        __syncthreads();
        v16h af[4], bf[2];
#pragma unroll
        for (int i = 0; i < 4; i++) af[i] = load_fragA(&sA[cur][wm * 64 + i * 16][0], 40, lane);
#pragma unroll
        for (int j = 0; j < 2; j++) bf[j] = load_fragB(&sB[cur][wn * 32 + j * 16][0], 40, lane);
#pragma unroll
        for (int i = 0; i < 4; i++)
#pragma unroll
            for (int j = 0; j < 2; j++) acc[i][j] = WMMA_F16(af[i], bf[j], acc[i][j]);
        __syncthreads();
    }

    const int mh = (lane >> 4) << 3, nn = lane & 15;
#pragma unroll
    for (int i = 0; i < 4; i++)
#pragma unroll
        for (int j = 0; j < 2; j++) {
            int gn = n0 + wn * 32 + j * 16 + nn;
            float bv = pb[gn];
#pragma unroll
            for (int r = 0; r < 8; r++) {
                int gm = m0 + wm * 64 + i * 16 + r + mh;
                out[(size_t)gm * 1024 + gn] = acc[i][j][r] + bv;
            }
        }
}

extern "C" void kernel_launch(void* const* d_in, const int* in_sizes, int n_in,
                              void* d_out, int out_size, void* d_ws, size_t ws_size,
                              hipStream_t stream) {
    const float* x      = (const float*)d_in[0];
    const float* qkv_w  = (const float*)d_in[1];
    const float* qkv_b  = (const float*)d_in[2];
    const float* proj_w = (const float*)d_in[3];
    const float* proj_b = (const float*)d_in[4];
    const float* lqa    = (const float*)d_in[5];
    const float* lqb    = (const float*)d_in[6];
    const float* lva    = (const float*)d_in[7];
    const float* lvb    = (const float*)d_in[8];
    float* out = (float*)d_out;

    // workspace layout (f16): x_h | qkv_w fused | proj_w | Q | K | V | attn_out  (~88 MB)
    _Float16* xh   = (_Float16*)d_ws;
    _Float16* wf   = xh   + (size_t)8 * 1024 * 1024;
    _Float16* pwh  = wf   + (size_t)3 * 1024 * 1024;
    _Float16* qws  = pwh  + (size_t)1024 * 1024;
    _Float16* kws  = qws  + (size_t)8 * 1024 * 1024;
    _Float16* vws  = kws  + (size_t)8 * 1024 * 1024;
    _Float16* ows2 = vws  + (size_t)8 * 1024 * 1024;

    k_cvt_x<<<8192, 256, 0, stream>>>(x, xh);
    k_fuse_w<<<12288, 256, 0, stream>>>(qkv_w, lqa, lqb, lva, lvb, proj_w, wf, pwh);
    k_qkv<<<dim3(24, 64), 256, 0, stream>>>(xh, wf, qkv_b, qws, kws, vws);
    k_attn<<<dim3(8, 128), 256, 0, stream>>>(qws, kws, vws, ows2);
    k_proj<<<dim3(8, 64), 256, 0, stream>>>(ows2, pwh, proj_b, out);
}
